// DeepGravityEasy_5076651344441
// MI455X (gfx1250) — matrix operand order repaired and checked by reading, not compiled, not run on hardware
//
#include <hip/hip_runtime.h>
#include <hip/hip_bf16.h>

typedef __attribute__((ext_vector_type(16))) _Float16 v16h;
typedef __attribute__((ext_vector_type(8)))  _Float16 v8h;
typedef __attribute__((ext_vector_type(8)))  float    v8f;
typedef __attribute__((ext_vector_type(4)))  float    v4f;

#define NUM_ORIGINS 4096

// Order-preserving float -> uint key so we can use hardware integer atomic max.
__device__ __forceinline__ unsigned fkey(float f) {
  unsigned b = __float_as_uint(f);
  return (b & 0x80000000u) ? ~b : (b | 0x80000000u);
}
__device__ __forceinline__ float funkey(unsigned k) {
  return __uint_as_float((k & 0x80000000u) ? (k & 0x7fffffffu) : ~k);
}

__global__ void dg_init(unsigned* __restrict__ segMaxBits, float* __restrict__ segSum) {
  int i = blockIdx.x * blockDim.x + threadIdx.x;
  if (i < NUM_ORIGINS) {
    segMaxBits[i] = fkey(-__builtin_inff());
    segSum[i] = 0.f;
  }
}

// Load a 16x64 f32 row tile into two f16 A operands (16x32 each).
// A layout: lane<16 -> M=lane, K runs {0..7, 16..23}; lane>=16 -> M=lane-16,
// K runs {8..15, 24..31}; kbA = 0 or 8 selects the sub-run.
__device__ __forceinline__ void loadA_x(const float* __restrict__ xr, int kbA, v16h& a0, v16h& a1) {
#pragma unroll
  for (int o = 0; o < 2; ++o) {
    const float* p1 = xr + o * 32 + kbA;
    const float* p2 = xr + o * 32 + 16 + kbA;
    v4f f0 = *(const v4f*)(p1);
    v4f f1 = *(const v4f*)(p1 + 4);
    v4f f2 = *(const v4f*)(p2);
    v4f f3 = *(const v4f*)(p2 + 4);
    v16h a;
#pragma unroll
    for (int h = 0; h < 4; ++h) {
      a[h]      = (_Float16)f0[h];
      a[4 + h]  = (_Float16)f1[h];
      a[8 + h]  = (_Float16)f2[h];
      a[12 + h] = (_Float16)f3[h];
    }
    if (o == 0) a0 = a; else a1 = a;
  }
}

// Weight slab in LDS: per (layer, operand, lane) the 16 B-layout halves are
// contiguous (2x ds_load_b128), padded to 24 halves (48B) so the lane bank
// stride is 12 banks -> only 2-way conflicts.
#define WPAD 24

__device__ __forceinline__ v16h loadB_lds(const _Float16* __restrict__ p) {
  v8h lo = *(const v8h*)(p);
  v8h hi = *(const v8h*)(p + 8);
  v16h b;
#pragma unroll
  for (int h = 0; h < 8; ++h) { b[h] = lo[h]; b[8 + h] = hi[h]; }
  return b;
}

// hbuf row padded to 72 halves (144B = 36 banks) -> conflict-free b128 reads.
#define HPAD 72

__global__ void __launch_bounds__(256) dg_mlp(
    const float* __restrict__ x, const int* __restrict__ ids,
    const float* __restrict__ W1, const float* __restrict__ b1,
    const float* __restrict__ W2, const float* __restrict__ b2,
    const float* __restrict__ W3, const float* __restrict__ b3,
    float* __restrict__ logits, unsigned* __restrict__ segMaxBits, int M)
{
  __shared__ __align__(16) _Float16 wlds[2][8][32][WPAD]; // 24 KB: f16 weights, B layout
  __shared__ __align__(16) _Float16 hbuf[8][16][HPAD];    // 18 KB: per-wave h1 staging

  const int lane   = threadIdx.x & 31;
  const int laneLo = lane & 15;
  const int hi     = lane >> 4;        // which 16-lane half
  const int w      = threadIdx.x >> 5; // wave index in block
  const int kbA    = hi ? 8 : 0;       // A-layout K sub-base

  // ---- stage f16 weights into LDS in per-lane B-operand order ----
  {
    int sl = threadIdx.x & 31;   // lane slot
    int op = threadIdx.x >> 5;   // operand 0..7 : nb = op>>1, o = op&1
    int n  = (op >> 1) * 16 + (sl & 15);
    int kB = (op & 1) * 32 + ((sl >> 4) ? 8 : 0);
#pragma unroll
    for (int h = 0; h < 16; ++h) {
      wlds[0][op][sl][h] = (_Float16)W1[(kB + h + (h >= 8 ? 8 : 0)) * 64 + n];
      wlds[1][op][sl][h] = (_Float16)W2[(kB + h + (h >= 8 ? 8 : 0)) * 64 + n];
    }
  }
  __syncthreads();

  float b1v[4], b2v[4], w3v[4];
#pragma unroll
  for (int nb = 0; nb < 4; ++nb) {
    int n = nb * 16 + laneLo;
    b1v[nb] = b1[n];
    b2v[nb] = b2[n];
    w3v[nb] = W3[n];
  }
  const float b3s = b3[0];

  const int nTiles = M >> 4;
  const int gw     = (blockIdx.x * blockDim.x + threadIdx.x) >> 5;
  const int stride = (gridDim.x * blockDim.x) >> 5;

  int tile = gw;
  v16h a0 = {}, a1 = {};
  if (tile < nTiles) loadA_x(x + (tile * 16 + laneLo) * 64, kbA, a0, a1);

  for (; tile < nTiles; tile += stride) {
    // prefetch next tile's A operands (overlaps with this tile's compute)
    int nxt = tile + stride;
    v16h na0 = a0, na1 = a1;
    if (nxt < nTiles) loadA_x(x + (nxt * 16 + laneLo) * 64, kbA, na0, na1);

    // ---- layer 1: h1 = relu(x @ W1 + b1), staged to LDS as f16 ----
#pragma unroll
    for (int nb = 0; nb < 4; ++nb) {
      v8f c;
#pragma unroll
      for (int i = 0; i < 8; ++i) c[i] = b1v[nb];  // C init = bias (N = lane)
      v16h B0 = loadB_lds(&wlds[0][nb * 2 + 0][lane][0]);
      v16h B1 = loadB_lds(&wlds[0][nb * 2 + 1][lane][0]);
      c = __builtin_amdgcn_wmma_f32_16x16x32_f16(false, a0, false, B0, (short)0, c, false, false);
      c = __builtin_amdgcn_wmma_f32_16x16x32_f16(false, a1, false, B1, (short)0, c, false, false);
#pragma unroll
      for (int r = 0; r < 8; ++r) {
        float v = c[r];
        v = v > 0.f ? v : 0.f;
        hbuf[w][r + (hi ? 8 : 0)][nb * 16 + laneLo] = (_Float16)v;  // C layout -> row-major
      }
    }

    // cross-lane handoff through LDS within the wave (DS is in-order per wave;
    // the clobber also stops the compiler from caching/hoisting LDS weights)
    asm volatile("s_wait_dscnt 0x0" ::: "memory");

    // ---- gather layer-2 A operands from LDS (row m = laneLo) ----
    v16h A0, A1;
#pragma unroll
    for (int o = 0; o < 2; ++o) {
      v8h r1 = *(const v8h*)&hbuf[w][laneLo][o * 32 + kbA];
      v8h r2 = *(const v8h*)&hbuf[w][laneLo][o * 32 + 16 + kbA];
      v16h a;
#pragma unroll
      for (int h = 0; h < 8; ++h) { a[h] = r1[h]; a[8 + h] = r2[h]; }
      if (o == 0) A0 = a; else A1 = a;
    }

    // ---- layer 2 + layer 3 partials: relu(h1 @ W2 + b2) dot W3 ----
    float acc[8];
#pragma unroll
    for (int r = 0; r < 8; ++r) acc[r] = 0.f;
#pragma unroll
    for (int nb = 0; nb < 4; ++nb) {
      v8f c;
#pragma unroll
      for (int i = 0; i < 8; ++i) c[i] = b2v[nb];
      v16h B0 = loadB_lds(&wlds[1][nb * 2 + 0][lane][0]);
      v16h B1 = loadB_lds(&wlds[1][nb * 2 + 1][lane][0]);
      c = __builtin_amdgcn_wmma_f32_16x16x32_f16(false, A0, false, B0, (short)0, c, false, false);
      c = __builtin_amdgcn_wmma_f32_16x16x32_f16(false, A1, false, B1, (short)0, c, false, false);
#pragma unroll
      for (int r = 0; r < 8; ++r) {
        float v = c[r];
        v = v > 0.f ? v : 0.f;
        acc[r] += v * w3v[nb];  // lane holds n = nb*16+laneLo -> multiply by W3[n]
      }
    }

    // reduce over N within each 16-lane half (xor masks < 16 stay in-half)
#pragma unroll
    for (int r = 0; r < 8; ++r) {
      acc[r] += __shfl_xor(acc[r], 1, 32);
      acc[r] += __shfl_xor(acc[r], 2, 32);
      acc[r] += __shfl_xor(acc[r], 4, 32);
      acc[r] += __shfl_xor(acc[r], 8, 32);
    }

    if (laneLo == 0) {  // lanes 0 and 16 each own 8 rows
      int mBase = tile * 16 + (hi ? 8 : 0);
#pragma unroll
      for (int r = 0; r < 8; ++r) {
        float lg = acc[r] + b3s;
        int m = mBase + r;
        logits[m] = lg;
        __hip_atomic_fetch_max(&segMaxBits[ids[m]], fkey(lg),
                               __ATOMIC_RELAXED, __HIP_MEMORY_SCOPE_AGENT);
      }
    }

    a0 = na0; a1 = na1;
  }
}

__global__ void dg_expsum(float* __restrict__ out, const int* __restrict__ ids,
                          const unsigned* __restrict__ segMaxBits,
                          float* __restrict__ segSum, int M) {
  int i = blockIdx.x * blockDim.x + threadIdx.x;
  int stride = gridDim.x * blockDim.x;
  for (; i < M; i += stride) {
    int id = ids[i];
    float e = __expf(out[i] - funkey(segMaxBits[id]));
    out[i] = e;
    __hip_atomic_fetch_add(&segSum[id], e, __ATOMIC_RELAXED, __HIP_MEMORY_SCOPE_AGENT);
  }
}

__global__ void dg_div(float* __restrict__ out, const int* __restrict__ ids,
                       const float* __restrict__ segSum, int M) {
  int i = blockIdx.x * blockDim.x + threadIdx.x;
  int stride = gridDim.x * blockDim.x;
  for (; i < M; i += stride) out[i] = out[i] / segSum[ids[i]];
}

extern "C" void kernel_launch(void* const* d_in, const int* in_sizes, int n_in,
                              void* d_out, int out_size, void* d_ws, size_t ws_size,
                              hipStream_t stream) {
  const float* x  = (const float*)d_in[0];
  const int*  ids = (const int*)d_in[1];
  const float* W1 = (const float*)d_in[2];
  const float* b1 = (const float*)d_in[3];
  const float* W2 = (const float*)d_in[4];
  const float* b2 = (const float*)d_in[5];
  const float* W3 = (const float*)d_in[6];
  const float* b3 = (const float*)d_in[7];
  float* out = (float*)d_out;
  const int M = in_sizes[0] / 64;

  unsigned* segMaxBits = (unsigned*)d_ws;
  float*    segSum     = (float*)((char*)d_ws + NUM_ORIGINS * sizeof(unsigned));

  dg_init<<<(NUM_ORIGINS + 255) / 256, 256, 0, stream>>>(segMaxBits, segSum);
  dg_mlp<<<1024, 256, 0, stream>>>(x, ids, W1, b1, W2, b2, W3, b3, out, segMaxBits, M);
  dg_expsum<<<2048, 256, 0, stream>>>(out, ids, segMaxBits, segSum, M);
  dg_div<<<2048, 256, 0, stream>>>(out, ids, segSum, M);
}
